// PriorFusion3D_voxel_17119739642230
// MI455X (gfx1250) — matrix operand order repaired and checked by reading, not compiled, not run on hardware
//
#include <hip/hip_runtime.h>

// ---------------------------------------------------------------------------
// Types for CDNA5 WMMA (wave32, 16x16x32 bf16 -> f32)
// ---------------------------------------------------------------------------
typedef __attribute__((ext_vector_type(16))) __bf16        bf16x16;
typedef __attribute__((ext_vector_type(8)))  float         f32x8;
typedef __attribute__((ext_vector_type(4)))  unsigned int  u32x4;

union Frag {
    bf16x16 v;
    u32x4   u[2];
};

__device__ inline unsigned short to_bf16(float x) {
    union { float f; unsigned u; } c; c.f = x;
    unsigned r = c.u + 0x7FFFu + ((c.u >> 16) & 1u);   // round-to-nearest-even
    return (unsigned short)(r >> 16);
}
__device__ inline float from_bf16(unsigned short h) {
    union { unsigned u; float f; } c; c.u = ((unsigned)h) << 16;
    return c.f;
}

#define BN_INV 0.9999950000374997f   // 1/sqrt(1+1e-5)

// ---------------------------------------------------------------------------
// K0: convert / rearrange weights to bf16 once per call
//   c1w (256x512)            -> w1b [o][k]
//   c2w (256,256,3,3)        -> w2b [dy*3+dx][o][i]
//   c3w (80x112)             -> w3b [o][128] (zero-padded K)
// ---------------------------------------------------------------------------
__global__ void __launch_bounds__(256) k_prep(
    const float* __restrict__ c1w, const float* __restrict__ c2w,
    const float* __restrict__ c3w,
    unsigned short* __restrict__ w1b, unsigned short* __restrict__ w2b,
    unsigned short* __restrict__ w3b)
{
    int i = blockIdx.x * 256 + threadIdx.x;
    if (i < 256 * 512) w1b[i] = to_bf16(c1w[i]);
    if (i < 9 * 256 * 256) {
        int ii = i & 255;
        int o  = (i >> 8) & 255;
        int f  = i >> 16;                      // dy*3+dx
        w2b[i] = to_bf16(c2w[o * 2304 + ii * 9 + f]);
    }
    if (i < 80 * 128) {
        int k = i & 127, o = i >> 7;
        w3b[i] = (k < 112) ? to_bf16(c3w[o * 112 + k]) : (unsigned short)0;
    }
}

// ---------------------------------------------------------------------------
// K1: per-point MLP (68->64 relu, 64->64 relu) + scatter into dense grid
//     vox layout: [y*400+x][512] bf16, channel = c*8 + z
// ---------------------------------------------------------------------------
__global__ void __launch_bounds__(256) k_mlp_scatter(
    const float* __restrict__ feats, const int* __restrict__ coords,
    const float* __restrict__ W1, const float* __restrict__ b1,
    const float* __restrict__ W2, const float* __restrict__ b2,
    unsigned short* __restrict__ vox, int N)
{
    __shared__ float sW1[68 * 64];
    __shared__ float sW2[64 * 64];
    __shared__ float sb1[64], sb2[64];
    for (int i = threadIdx.x; i < 68 * 64; i += 256) sW1[i] = W1[i];
    for (int i = threadIdx.x; i < 64 * 64; i += 256) sW2[i] = W2[i];
    if (threadIdx.x < 64) { sb1[threadIdx.x] = b1[threadIdx.x];
                            sb2[threadIdx.x] = b2[threadIdx.x]; }
    __syncthreads();

    int n = blockIdx.x * 256 + threadIdx.x;
    if (n >= N) return;

    float f[68];
#pragma unroll
    for (int j = 0; j < 68; ++j) f[j] = feats[(size_t)n * 68 + j];

    float h[64];
#pragma unroll
    for (int c = 0; c < 64; ++c) {
        float a = sb1[c];
#pragma unroll
        for (int j = 0; j < 68; ++j) a += f[j] * sW1[j * 64 + c];
        h[c] = fmaxf(a, 0.f);
    }

    int cx = coords[n * 3 + 0];
    int cy = coords[n * 3 + 1];
    int cz = coords[n * 3 + 2];
    size_t base = ((size_t)(cy * 400 + cx)) * 512u + (unsigned)cz;

#pragma unroll
    for (int c = 0; c < 64; ++c) {
        float a = sb2[c];
#pragma unroll
        for (int k = 0; k < 64; ++k) a += h[k] * sW2[k * 64 + c];
        vox[base + (size_t)c * 8u] = to_bf16(fmaxf(a, 0.f));
    }
}

// ---------------------------------------------------------------------------
// K2: conv1 1x1 (512->256) + bias + BN + ReLU,  WMMA bf16
//     block = 8 waves, 16-pixel tile; each wave -> 2 channel tiles (all 256)
// ---------------------------------------------------------------------------
__global__ void __launch_bounds__(256) k_conv1(
    const unsigned short* __restrict__ vox,     // [160000][512]
    const unsigned short* __restrict__ w1b,     // [256][512]
    const float* __restrict__ c1b, const float* __restrict__ bn1g,
    const float* __restrict__ bn1b,
    unsigned short* __restrict__ p1)            // [160000][256]
{
    __shared__ unsigned short sB[16 * 512];     // 16 KB, contiguous tile
    const int tid     = threadIdx.x;
    const int pixBase = blockIdx.x * 16;
    {
        const u32x4* src = (const u32x4*)(vox + (size_t)pixBase * 512);
        u32x4*       dst = (u32x4*)sB;
        for (int i = tid; i < 1024; i += 256) dst[i] = src[i];
    }
    __syncthreads();

    const int lane = tid & 31, wave = tid >> 5;
    const int n = lane & 15, half = lane >> 4;
    const int ct0 = wave * 2, ct1 = ct0 + 1;

    f32x8 acc0 = {}, acc1 = {};
    const unsigned short* a0row = w1b + (size_t)(ct0 * 16 + n) * 512;
    const unsigned short* a1row = w1b + (size_t)(ct1 * 16 + n) * 512;

    for (int kk = 0; kk < 512; kk += 32) {
        Frag b, a0, a1;
        const u32x4* bp = (const u32x4*)(sB + n * 512 + kk + half * 8);
        b.u[0] = bp[0]; b.u[1] = bp[2];
        a0.u[0] = *(const u32x4*)(a0row + kk + half * 8);
        a0.u[1] = *(const u32x4*)(a0row + kk + half * 8 + 16);
        a1.u[0] = *(const u32x4*)(a1row + kk + half * 8);
        a1.u[1] = *(const u32x4*)(a1row + kk + half * 8 + 16);
        acc0 = __builtin_amdgcn_wmma_f32_16x16x32_bf16(false, a0.v, false, b.v,
                                                       (short)0, acc0, false, false);
        acc1 = __builtin_amdgcn_wmma_f32_16x16x32_bf16(false, a1.v, false, b.v,
                                                       (short)0, acc1, false, false);
    }

    const size_t pixOff = (size_t)(pixBase + n) * 256;
#pragma unroll
    for (int t = 0; t < 2; ++t) {
        f32x8 acc = t ? acc1 : acc0;
        int chBase = (t ? ct1 : ct0) * 16 + half * 8;
        union { unsigned short s[8]; u32x4 v; } o;
#pragma unroll
        for (int r = 0; r < 8; ++r) {
            int ch = chBase + r;
            float v = acc[r] + c1b[ch];
            v = v * (bn1g[ch] * BN_INV) + bn1b[ch];
            o.s[r] = to_bf16(fmaxf(v, 0.f));
        }
        *(u32x4*)(p1 + pixOff + chBase) = o.v;
    }
}

// ---------------------------------------------------------------------------
// K3: conv2 3x3 SAME (256->256) + bias + BN + ReLU, WMMA bf16
//     grid (25, 400): 16-pixel row tile; LDS halo [3][18][256] bf16
// ---------------------------------------------------------------------------
__global__ void __launch_bounds__(256) k_conv2(
    const unsigned short* __restrict__ p1,      // [160000][256]
    const unsigned short* __restrict__ w2b,     // [9][256][256]
    const float* __restrict__ c2b, const float* __restrict__ bn2g,
    const float* __restrict__ bn2b,
    unsigned short* __restrict__ p2)            // [160000][256]
{
    __shared__ unsigned short sB[3 * 18 * 256]; // 27648 B
    const int tid   = threadIdx.x;
    const int xbase = blockIdx.x * 16;
    const int y     = blockIdx.y;

    for (int i = tid; i < 1728; i += 256) {     // 54 pixels * 32 u32x4
        int chunk = i & 31;
        int px    = i >> 5;
        int r  = px / 18;
        int x  = px % 18;
        int gy = y + r - 1;
        int gx = xbase + x - 1;
        u32x4 val = {};
        if (gy >= 0 && gy < 400 && gx >= 0 && gx < 400)
            val = ((const u32x4*)(p1 + (size_t)(gy * 400 + gx) * 256))[chunk];
        ((u32x4*)sB)[i] = val;
    }
    __syncthreads();

    const int lane = tid & 31, wave = tid >> 5;
    const int n = lane & 15, half = lane >> 4;
    const int ct0 = wave * 2, ct1 = ct0 + 1;
    f32x8 acc0 = {}, acc1 = {};

    for (int f = 0; f < 9; ++f) {
        const int dy = f / 3, dx = f % 3;
        const unsigned short* sBp   = sB + (dy * 18 + n + dx) * 256 + half * 8;
        const unsigned short* a0row = w2b + ((size_t)f * 256 + ct0 * 16 + n) * 256;
        const unsigned short* a1row = w2b + ((size_t)f * 256 + ct1 * 16 + n) * 256;
        for (int kk = 0; kk < 256; kk += 32) {
            Frag b, a0, a1;
            b.u[0] = *(const u32x4*)(sBp + kk);
            b.u[1] = *(const u32x4*)(sBp + kk + 16);
            a0.u[0] = *(const u32x4*)(a0row + kk + half * 8);
            a0.u[1] = *(const u32x4*)(a0row + kk + half * 8 + 16);
            a1.u[0] = *(const u32x4*)(a1row + kk + half * 8);
            a1.u[1] = *(const u32x4*)(a1row + kk + half * 8 + 16);
            acc0 = __builtin_amdgcn_wmma_f32_16x16x32_bf16(false, a0.v, false, b.v,
                                                           (short)0, acc0, false, false);
            acc1 = __builtin_amdgcn_wmma_f32_16x16x32_bf16(false, a1.v, false, b.v,
                                                           (short)0, acc1, false, false);
        }
    }

    const size_t pixOff = (size_t)(y * 400 + xbase + n) * 256;
#pragma unroll
    for (int t = 0; t < 2; ++t) {
        f32x8 acc = t ? acc1 : acc0;
        int chBase = (t ? ct1 : ct0) * 16 + half * 8;
        union { unsigned short s[8]; u32x4 v; } o;
#pragma unroll
        for (int r = 0; r < 8; ++r) {
            int ch = chBase + r;
            float v = acc[r] + c2b[ch];
            v = v * (bn2g[ch] * BN_INV) + bn2b[ch];
            o.s[r] = to_bf16(fmaxf(v, 0.f));
        }
        *(u32x4*)(p2 + pixOff + chBase) = o.v;
    }
}

// ---------------------------------------------------------------------------
// K4: 2x2 max pool (400x400 -> 200x200), bf16, [pix][256] layout
// ---------------------------------------------------------------------------
__global__ void __launch_bounds__(256) k_pool(
    const unsigned short* __restrict__ p2, unsigned short* __restrict__ pooled)
{
    int i = blockIdx.x * 256 + threadIdx.x;
    if (i >= 40000 * 256) return;
    int ch  = i & 255;
    int pix = i >> 8;
    int w = pix % 200, h = pix / 200;
    const unsigned short* b00 = p2 + ((size_t)(2 * h * 400 + 2 * w)) * 256 + ch;
    float v = from_bf16(b00[0]);
    v = fmaxf(v, from_bf16(b00[256]));
    v = fmaxf(v, from_bf16(b00[400 * 256]));
    v = fmaxf(v, from_bf16(b00[400 * 256 + 256]));
    pooled[i] = to_bf16(v);
}

// ---------------------------------------------------------------------------
// K5: build xcat [pos][128] bf16 : ch<80 = bev, 80..111 = pooled, 112..127 = 0
//     pos = (h*200+w)*8 + z
// ---------------------------------------------------------------------------
__global__ void __launch_bounds__(256) k_xcat(
    const float* __restrict__ bev, const unsigned short* __restrict__ pooled,
    unsigned short* __restrict__ xcat)
{
    long i = (long)blockIdx.x * 256 + threadIdx.x;
    if (i >= 40960000L) return;
    int ch  = (int)(i & 127);
    int pos = (int)(i >> 7);
    unsigned short v = 0;
    if (ch < 80) {
        v = to_bf16(bev[(size_t)ch * 320000 + pos]);
    } else if (ch < 112) {
        int z = pos & 7, hw = pos >> 3;
        v = pooled[(size_t)hw * 256 + (ch - 80) * 8 + z];
    }
    xcat[i] = v;
}

// ---------------------------------------------------------------------------
// K6: conv3 1x1x1 (112->80, K padded to 128) + bias + BN + residual + ReLU
//     grid 20000 x 160 threads (5 waves = 5 output-channel tiles)
// ---------------------------------------------------------------------------
__global__ void __launch_bounds__(160) k_conv3(
    const unsigned short* __restrict__ xcat,    // [320000][128]
    const unsigned short* __restrict__ w3b,     // [80][128]
    const float* __restrict__ c3b, const float* __restrict__ bn3g,
    const float* __restrict__ bn3b,
    const float* __restrict__ bev, float* __restrict__ out)
{
    __shared__ unsigned short sB[16 * 128];     // 4 KB
    const int tid     = threadIdx.x;
    const int pixBase = blockIdx.x * 16;
    {
        const u32x4* src = (const u32x4*)(xcat + (size_t)pixBase * 128);
        u32x4*       dst = (u32x4*)sB;
        for (int i = tid; i < 256; i += 160) dst[i] = src[i];
    }
    __syncthreads();

    const int lane = tid & 31, wave = tid >> 5;   // wave = channel tile 0..4
    const int n = lane & 15, half = lane >> 4;

    f32x8 acc = {};
    const unsigned short* arow = w3b + (size_t)(wave * 16 + n) * 128;
    for (int kk = 0; kk < 128; kk += 32) {
        Frag a, b;
        const unsigned short* bp = sB + n * 128 + kk + half * 8;
        b.u[0] = *(const u32x4*)bp;
        b.u[1] = *(const u32x4*)(bp + 16);
        a.u[0] = *(const u32x4*)(arow + kk + half * 8);
        a.u[1] = *(const u32x4*)(arow + kk + half * 8 + 16);
        acc = __builtin_amdgcn_wmma_f32_16x16x32_bf16(false, a.v, false, b.v,
                                                      (short)0, acc, false, false);
    }

    const int pix = pixBase + n;
#pragma unroll
    for (int r = 0; r < 8; ++r) {
        int o = wave * 16 + half * 8 + r;         // < 80 always
        float v = acc[r] + c3b[o];
        v = v * (bn3g[o] * BN_INV) + bn3b[o];
        v += bev[(size_t)o * 320000 + pix];
        out[(size_t)o * 320000 + pix] = fmaxf(v, 0.f);
    }
}

// ---------------------------------------------------------------------------
// Launch
// ---------------------------------------------------------------------------
extern "C" void kernel_launch(void* const* d_in, const int* in_sizes, int n_in,
                              void* d_out, int out_size, void* d_ws, size_t ws_size,
                              hipStream_t stream)
{
    const float* bev  = (const float*)d_in[0];
    const float* pf   = (const float*)d_in[1];
    const int*   pc   = (const int*)  d_in[2];
    const float* W1   = (const float*)d_in[3];
    const float* b1   = (const float*)d_in[4];
    const float* W2   = (const float*)d_in[5];
    const float* b2   = (const float*)d_in[6];
    const float* c1w  = (const float*)d_in[7];
    const float* c1b  = (const float*)d_in[8];
    const float* bn1g = (const float*)d_in[9];
    const float* bn1b = (const float*)d_in[10];
    const float* c2w  = (const float*)d_in[11];
    const float* c2b  = (const float*)d_in[12];
    const float* bn2g = (const float*)d_in[13];
    const float* bn2b = (const float*)d_in[14];
    const float* c3w  = (const float*)d_in[15];
    const float* c3b  = (const float*)d_in[16];
    const float* bn3g = (const float*)d_in[17];
    const float* bn3b = (const float*)d_in[18];

    char* ws = (char*)d_ws;
    unsigned short* vox    = (unsigned short*)(ws + 0);          // 163,840,000 B
    unsigned short* p1     = (unsigned short*)(ws + 163840000);  //  81,920,000 B
    unsigned short* pooled = (unsigned short*)(ws + 245760000);  //  20,480,000 B
    unsigned short* xcat   = (unsigned short*)(ws + 266240000);  //  81,920,000 B
    unsigned short* w1b    = (unsigned short*)(ws + 348160000);  //     262,144 B
    unsigned short* w2b    = (unsigned short*)(ws + 348422144);  //   1,179,648 B
    unsigned short* w3b    = (unsigned short*)(ws + 349601792);  //      20,480 B
    unsigned short* p2     = vox;  // vox dead after conv1; reuse for conv2 out

    hipMemsetAsync(vox, 0, 163840000, stream);
    k_prep<<<(589824 + 255) / 256, 256, 0, stream>>>(c1w, c2w, c3w, w1b, w2b, w3b);
    k_mlp_scatter<<<(200000 + 255) / 256, 256, 0, stream>>>(pf, pc, W1, b1, W2, b2,
                                                            vox, 200000);
    k_conv1<<<10000, 256, 0, stream>>>(vox, w1b, c1b, bn1g, bn1b, p1);
    k_conv2<<<dim3(25, 400), 256, 0, stream>>>(p1, w2b, c2b, bn2g, bn2b, p2);
    k_pool<<<(40000 * 256 + 255) / 256, 256, 0, stream>>>(p2, pooled);
    k_xcat<<<(40960000 + 255) / 256, 256, 0, stream>>>(bev, pooled, xcat);
    k_conv3<<<20000, 160, 0, stream>>>(xcat, w3b, c3b, bn3g, bn3b, bev, (float*)d_out);
}
